// BAttn_13486197309986
// MI455X (gfx1250) — matrix operand order repaired and checked by reading, not compiled, not run on hardware
//
#include <hip/hip_runtime.h>

typedef __attribute__((ext_vector_type(16))) __bf16 v16bf;
typedef __attribute__((ext_vector_type(8)))  __bf16 v8bf;
typedef __attribute__((ext_vector_type(8)))  float  v8f;

#define B_  2
#define T_  2048
#define D_  1024
#define H_  16
#define HD_ 64
#define BT_ (B_ * T_)      // 4096
#define N3_ (3 * D_)       // 3072

__device__ __forceinline__ v8f wmma_bf16(v16bf a, v16bf b, v8f c) {
    // D = A(16x32 bf16) x B(32x16 bf16) + C(16x16 f32)
    return __builtin_amdgcn_wmma_f32_16x16x32_bf16(
        /*neg_a=*/false, a, /*neg_b=*/false, b,
        /*c_mod=*/(short)0, c, /*reuse_a=*/false, /*reuse_b=*/false);
}

// A-fragment (16x32 bf16): lane row = l16, element e -> K = half*16+e
__device__ __forceinline__ v16bf load_a_frag(const __bf16* rowp, int k0, int half) {
    v8bf lo = *(const v8bf*)(rowp + k0 + half * 16);
    v8bf hi = *(const v8bf*)(rowp + k0 + half * 16 + 8);
    v16bf a;
#pragma unroll
    for (int i = 0; i < 8; ++i) { a[i] = lo[i]; a[i + 8] = hi[i]; }
    return a;
}

__device__ __forceinline__ float rmax16(float v) {
    v = fmaxf(v, __shfl_xor(v, 1));
    v = fmaxf(v, __shfl_xor(v, 2));
    v = fmaxf(v, __shfl_xor(v, 4));
    v = fmaxf(v, __shfl_xor(v, 8));
    return v;
}
__device__ __forceinline__ float rsum16(float v) {
    v += __shfl_xor(v, 1);
    v += __shfl_xor(v, 2);
    v += __shfl_xor(v, 4);
    v += __shfl_xor(v, 8);
    return v;
}

// ---------------- prep kernels ----------------
__global__ void k_f32_to_bf16(const float* __restrict__ in, __bf16* __restrict__ out, int n) {
    int i = blockIdx.x * blockDim.x + threadIdx.x;
    if (i < n) out[i] = (__bf16)in[i];
}

// in[K][N] (f32, row-major) -> out[N][K] (bf16)
__global__ void k_transpose_bf16(const float* __restrict__ in, __bf16* __restrict__ out,
                                 int K, int N) {
    int i = blockIdx.x * blockDim.x + threadIdx.x;
    if (i < K * N) {
        int k = i / N, n = i % N;
        out[(size_t)n * K + k] = (__bf16)in[i];
    }
}

// ---------------- QKV projection (64x64 register-blocked wave tile) ----------------
// xb[BT][D] bf16  x  wt[N3][D] bf16 (pre-transposed)  (+bias)
// scatter: Q -> qb[b][h][t][d], K -> kb[b][h][t][d], V -> vt[b][h][d][t]
__global__ __launch_bounds__(128) void k_qkv_gemm(
    const __bf16* __restrict__ xb, const __bf16* __restrict__ wt,
    const float* __restrict__ bqkv,
    __bf16* __restrict__ qb, __bf16* __restrict__ kb, __bf16* __restrict__ vt)
{
    const int lane = threadIdx.x & 31, wid = threadIdx.x >> 5;
    const int half = lane >> 4, l16 = lane & 15;
    const int tm = blockIdx.x;                 // 0..63   (BT/64)
    const int tn = blockIdx.y * 4 + wid;       // 0..47   (N3/64)

    const __bf16* a_base = xb + ((size_t)tm * 64 + l16) * D_;
    const __bf16* b_base = wt + ((size_t)tn * 64 + l16) * D_;

    v8f acc[4][4] = {};
#pragma unroll 1
    for (int k0 = 0; k0 < D_; k0 += 32) {
        v16bf A[4], Bv[4];
#pragma unroll
        for (int i = 0; i < 4; ++i)
            A[i] = load_a_frag(a_base + (size_t)i * 16 * D_, k0, half);
#pragma unroll
        for (int j = 0; j < 4; ++j)
            Bv[j] = *(const v16bf*)(b_base + (size_t)j * 16 * D_ + k0 + half * 16);
        __builtin_prefetch(a_base + k0 + 512, 0, 3);   // gfx1250 global_prefetch_b8
        __builtin_prefetch(b_base + k0 + 512, 0, 3);
#pragma unroll
        for (int i = 0; i < 4; ++i)
#pragma unroll
            for (int j = 0; j < 4; ++j)
                acc[i][j] = wmma_bf16(A[i], Bv[j], acc[i][j]);
    }

#pragma unroll
    for (int j = 0; j < 4; ++j) {
        const int ncol = tn * 64 + j * 16 + l16;
        const float bias = bqkv[ncol];
        const int s   = ncol >> 10;       // which of Q/K/V (uniform per 16-col tile)
        const int rem = ncol & 1023;
        const int hh  = rem >> 6;
        const int d   = rem & 63;
#pragma unroll
        for (int i = 0; i < 4; ++i) {
#pragma unroll
            for (int r = 0; r < 8; ++r) {
                const int bt = tm * 64 + i * 16 + r + 8 * half;
                const int bb = bt >> 11;      // / T_
                const int t  = bt & (T_ - 1);
                const float v = acc[i][j][r] + bias;
                const size_t rowmaj = (((size_t)bb * H_ + hh) * T_ + t) * HD_ + d;
                if (s == 0)      qb[rowmaj] = (__bf16)v;
                else if (s == 1) kb[rowmaj] = (__bf16)v;
                else vt[(((size_t)bb * H_ + hh) * HD_ + d) * T_ + t] = (__bf16)v;
            }
        }
    }
}

// ---------------- flash attention (32 queries / wave, K & V frags reused 2x) ----------------
__global__ __launch_bounds__(128) void k_attn(
    const __bf16* __restrict__ qb, const __bf16* __restrict__ kb,
    const __bf16* __restrict__ vt, __bf16* __restrict__ ctxb)
{
    __shared__ __align__(32) __bf16 Plds[4][2][16][32];   // 8 KB

    const int lane = threadIdx.x & 31, wid = threadIdx.x >> 5;
    const int half = lane >> 4, l16 = lane & 15;
    const int gw  = blockIdx.x * 4 + wid;      // 0..2047
    const int qt2 = gw & 63;                   // 32-query block (T/32)
    const int h   = (gw >> 6) & (H_ - 1);
    const int b   = gw >> 10;
    const size_t bh = (size_t)(b * H_ + h);

    // Q: two q-tiles, each two A fragments (K-dim = HD = 64)
    v16bf qa[2][2];
#pragma unroll
    for (int q = 0; q < 2; ++q) {
        const __bf16* qbase = qb + (bh * T_ + qt2 * 32 + q * 16 + l16) * HD_;
        qa[q][0] = load_a_frag(qbase, 0, half);
        qa[q][1] = load_a_frag(qbase, 32, half);
    }

    float mrow[2][8], lsum[2][8];
    v8f o[2][4] = {};
#pragma unroll
    for (int q = 0; q < 2; ++q)
#pragma unroll
        for (int r = 0; r < 8; ++r) { mrow[q][r] = -__builtin_inff(); lsum[q][r] = 0.0f; }

    const __bf16* kbase = kb + bh * (size_t)T_ * HD_;
    const __bf16* vbase = vt + bh * (size_t)HD_ * T_;

    for (int kt = 0; kt < T_; kt += 32) {
        // ---- K fragments for 32 keys (shared by both q-tiles) ----
        const __bf16* kp0 = kbase + (size_t)(kt + l16) * HD_;
        const __bf16* kp1 = kbase + (size_t)(kt + 16 + l16) * HD_;
        v16bf b00 = *(const v16bf*)(kp0 + half * 16);
        v16bf b01 = *(const v16bf*)(kp0 + 32 + half * 16);
        v16bf b10 = *(const v16bf*)(kp1 + half * 16);
        v16bf b11 = *(const v16bf*)(kp1 + 32 + half * 16);

        v8f s[2][2] = {};
#pragma unroll
        for (int q = 0; q < 2; ++q) {
            s[q][0] = wmma_bf16(qa[q][0], b00, s[q][0]);
            s[q][0] = wmma_bf16(qa[q][1], b01, s[q][0]);
            s[q][1] = wmma_bf16(qa[q][0], b10, s[q][1]);
            s[q][1] = wmma_bf16(qa[q][1], b11, s[q][1]);
        }

        // ---- online softmax over these 32 keys ----
#pragma unroll
        for (int q = 0; q < 2; ++q) {
#pragma unroll
            for (int r = 0; r < 8; ++r) {
                const float x0 = s[q][0][r] * 0.125f;   // 1/sqrt(64)
                const float x1 = s[q][1][r] * 0.125f;
                const float mx = rmax16(fmaxf(x0, x1));
                const float nm = fmaxf(mrow[q][r], mx);
                const float p0 = __expf(x0 - nm);
                const float p1 = __expf(x1 - nm);
                const float al = __expf(mrow[q][r] - nm);
                const float rs = rsum16(p0 + p1);
                lsum[q][r] = lsum[q][r] * al + rs;
                mrow[q][r] = nm;
                s[q][0][r] = p0; s[q][1][r] = p1;
                o[q][0][r] *= al; o[q][1][r] *= al;
                o[q][2][r] *= al; o[q][3][r] *= al;
            }
        }

        // ---- relayout P: D-fragment -> A-fragment via LDS ----
        __syncthreads();
#pragma unroll
        for (int q = 0; q < 2; ++q)
#pragma unroll
            for (int r = 0; r < 8; ++r) {
                const int m = r + 8 * half;
                Plds[wid][q][m][l16]      = (__bf16)s[q][0][r];
                Plds[wid][q][m][16 + l16] = (__bf16)s[q][1][r];
            }
        __syncthreads();
        v16bf pa[2];
#pragma unroll
        for (int q = 0; q < 2; ++q) {
            v8bf lo = *(const v8bf*)&Plds[wid][q][l16][half * 16];
            v8bf hi = *(const v8bf*)&Plds[wid][q][l16][half * 16 + 8];
#pragma unroll
            for (int i = 0; i < 8; ++i) { pa[q][i] = lo[i]; pa[q][i + 8] = hi[i]; }
        }

        // ---- ctx accumulation: P(16x32) x V(32x64), V frags shared by q-tiles ----
#pragma unroll
        for (int j = 0; j < 4; ++j) {
            const __bf16* vr = vbase + (size_t)(j * 16 + l16) * T_ + kt + half * 16;
            const v16bf vbf = *(const v16bf*)vr;
            o[0][j] = wmma_bf16(pa[0], vbf, o[0][j]);
            o[1][j] = wmma_bf16(pa[1], vbf, o[1][j]);
        }
    }

    // ---- normalize + write ctx (bf16, [b][t][h*64+d]) ----
#pragma unroll
    for (int q = 0; q < 2; ++q) {
        const size_t outbase = ((size_t)b * T_ + qt2 * 32 + q * 16) * D_ + h * HD_;
#pragma unroll
        for (int r = 0; r < 8; ++r) {
            const int m = r + 8 * half;
            const float inv = 1.0f / lsum[q][r];
            const size_t ob = outbase + (size_t)m * D_ + l16;
#pragma unroll
            for (int j = 0; j < 4; ++j)
                ctxb[ob + j * 16] = (__bf16)(o[q][j][r] * inv);
        }
    }
}

// ---------------- output projection (64x64 register-blocked wave tile) ----------------
__global__ __launch_bounds__(128) void k_out_gemm(
    const __bf16* __restrict__ ctxb, const __bf16* __restrict__ wt,
    const float* __restrict__ bout, float* __restrict__ out)
{
    const int lane = threadIdx.x & 31, wid = threadIdx.x >> 5;
    const int half = lane >> 4, l16 = lane & 15;
    const int tm = blockIdx.x;                 // 0..63
    const int tn = blockIdx.y * 4 + wid;       // 0..15  (D/64)

    const __bf16* a_base = ctxb + ((size_t)tm * 64 + l16) * D_;
    const __bf16* b_base = wt   + ((size_t)tn * 64 + l16) * D_;

    v8f acc[4][4] = {};
#pragma unroll 1
    for (int k0 = 0; k0 < D_; k0 += 32) {
        v16bf A[4], Bv[4];
#pragma unroll
        for (int i = 0; i < 4; ++i)
            A[i] = load_a_frag(a_base + (size_t)i * 16 * D_, k0, half);
#pragma unroll
        for (int j = 0; j < 4; ++j)
            Bv[j] = *(const v16bf*)(b_base + (size_t)j * 16 * D_ + k0 + half * 16);
        __builtin_prefetch(a_base + k0 + 512, 0, 3);
        __builtin_prefetch(b_base + k0 + 512, 0, 3);
#pragma unroll
        for (int i = 0; i < 4; ++i)
#pragma unroll
            for (int j = 0; j < 4; ++j)
                acc[i][j] = wmma_bf16(A[i], Bv[j], acc[i][j]);
    }

#pragma unroll
    for (int j = 0; j < 4; ++j) {
        const int ncol = tn * 64 + j * 16 + l16;
        const float bias = bout[ncol];
#pragma unroll
        for (int i = 0; i < 4; ++i) {
#pragma unroll
            for (int r = 0; r < 8; ++r) {
                const int row = tm * 64 + i * 16 + r + 8 * half;
                out[(size_t)row * D_ + ncol] = acc[i][j][r] + bias;
            }
        }
    }
}

extern "C" void kernel_launch(void* const* d_in, const int* in_sizes, int n_in,
                              void* d_out, int out_size, void* d_ws, size_t ws_size,
                              hipStream_t stream) {
    const float* x    = (const float*)d_in[0];   // [B,T,D]
    const float* Wqkv = (const float*)d_in[1];   // [D, 3*D]
    const float* bqkv = (const float*)d_in[2];   // [3*D]
    const float* Wout = (const float*)d_in[3];   // [D, D]
    const float* bout = (const float*)d_in[4];   // [D]
    float* out = (float*)d_out;                  // [B,T,D] f32

    char* ws = (char*)d_ws;
    size_t off = 0;
    __bf16* xb     = (__bf16*)(ws + off); off += (size_t)BT_ * D_ * 2;   // 8 MB
    __bf16* wqkvt  = (__bf16*)(ws + off); off += (size_t)N3_ * D_ * 2;   // 6 MB
    __bf16* woutt  = (__bf16*)(ws + off); off += (size_t)D_  * D_ * 2;   // 2 MB
    __bf16* qb     = (__bf16*)(ws + off); off += (size_t)BT_ * D_ * 2;   // 8 MB
    __bf16* kb     = (__bf16*)(ws + off); off += (size_t)BT_ * D_ * 2;   // 8 MB
    __bf16* vt     = (__bf16*)(ws + off); off += (size_t)BT_ * D_ * 2;   // 8 MB
    __bf16* ctxb   = (__bf16*)(ws + off); off += (size_t)BT_ * D_ * 2;   // 8 MB

    // prep
    {
        int n = BT_ * D_;
        k_f32_to_bf16<<<(n + 255) / 256, 256, 0, stream>>>(x, xb, n);
    }
    {
        int n = D_ * N3_;
        k_transpose_bf16<<<(n + 255) / 256, 256, 0, stream>>>(Wqkv, wqkvt, D_, N3_);
    }
    {
        int n = D_ * D_;
        k_transpose_bf16<<<(n + 255) / 256, 256, 0, stream>>>(Wout, woutt, D_, D_);
    }

    // qkv = x @ Wqkv + b  (scatter q/k/v; v transposed)
    k_qkv_gemm<<<dim3(BT_ / 64, N3_ / 256), 128, 0, stream>>>(xb, wqkvt, bqkv, qb, kb, vt);

    // flash attention: B*H*(T/32) waves, 4 waves/block
    k_attn<<<(B_ * H_ * (T_ / 32)) / 4, 128, 0, stream>>>(qb, kb, vt, ctxb);

    // out = ctx @ Wout + b
    k_out_gemm<<<dim3(BT_ / 64, D_ / 256), 128, 0, stream>>>(ctxb, woutt, bout, out);
}
